// LLaMAMoE_420906795085
// MI455X (gfx1250) — matrix operand order repaired and checked by reading, not compiled, not run on hardware
//
#include <hip/hip_runtime.h>
#include <hip/hip_bf16.h>
#include <math.h>

#define N_TOK 1024
#define C_DIM 2048
#define I_DIM 5632
#define E_NUM 8
#define MB    64          // tokens per block tile (4 WMMA sub-tiles)

typedef __attribute__((ext_vector_type(16))) __bf16 v16bf;
typedef __attribute__((ext_vector_type(2)))  __bf16 v2bf;
typedef __attribute__((ext_vector_type(8)))  float  v8f;

// ---------- bf16 conversion helpers (hardware packed cvt when available) ----
static __device__ __forceinline__ unsigned cvt_pk_bits(float a, float b) {
#if __has_builtin(__builtin_amdgcn_cvt_pk_bf16_f32)
    auto t = __builtin_amdgcn_cvt_pk_bf16_f32(a, b);
    unsigned u; __builtin_memcpy(&u, &t, 4); return u;
#else
    v2bf r; r[0] = (__bf16)a; r[1] = (__bf16)b;
    unsigned u; __builtin_memcpy(&u, &r, 4); return u;
#endif
}
static __device__ __forceinline__ unsigned short bfbits(float f) {
    __bf16 b = (__bf16)f;
    unsigned short s; __builtin_memcpy(&s, &b, 2); return s;
}

// ---------- 0: zero counters ----------
__global__ void moe_zero_cnt(int* cnt) {
    if (threadIdx.x < E_NUM) cnt[threadIdx.x] = 0;
}

// ---------- 1: router (one wave32 per token) ----------
__global__ __launch_bounds__(256)
void moe_router(const float* __restrict__ x, const float* __restrict__ Wg,
                int* __restrict__ cnt, int* __restrict__ tok,
                float* __restrict__ wrec) {
    const int wave = threadIdx.x >> 5;
    const int lane = threadIdx.x & 31;
    const int t = blockIdx.x * 8 + wave;
    if (t >= N_TOK) return;

    float p[E_NUM];
#pragma unroll
    for (int e = 0; e < E_NUM; ++e) p[e] = 0.0f;

    const float* xr = x + (size_t)t * C_DIM;
    for (int c = lane; c < C_DIM; c += 32) {
        float xv = xr[c];
#pragma unroll
        for (int e = 0; e < E_NUM; ++e) p[e] += xv * Wg[e * C_DIM + c];
    }
#pragma unroll
    for (int off = 16; off > 0; off >>= 1) {
#pragma unroll
        for (int e = 0; e < E_NUM; ++e) p[e] += __shfl_xor(p[e], off, 32);
    }

    if (lane == 0) {
        int e0 = 0; float v0 = p[0];
#pragma unroll
        for (int e = 1; e < E_NUM; ++e) if (p[e] > v0) { v0 = p[e]; e0 = e; }
        int e1 = -1; float v1 = -1e30f;
#pragma unroll
        for (int e = 0; e < E_NUM; ++e)
            if (e != e0 && (e1 < 0 || p[e] > v1)) { v1 = p[e]; e1 = e; }
        float w0 = 1.0f / (1.0f + __expf(v1 - v0));   // softmax of {v0,v1}
        float w1 = 1.0f - w0;
        int pos0 = atomicAdd(&cnt[e0], 1);
        tok[e0 * N_TOK + pos0] = t * 2;
        int pos1 = atomicAdd(&cnt[e1], 1);
        tok[e1 * N_TOK + pos1] = t * 2 + 1;
        wrec[t * 2]     = w0;
        wrec[t * 2 + 1] = w1;
    }
}

// ---------- 2: x fp32 -> bf16 ----------
__global__ __launch_bounds__(256)
void moe_cvt_x(const float* __restrict__ x, unsigned* __restrict__ xb32) {
    int idx = blockIdx.x * 256 + threadIdx.x;      // grid covers N*C/2 exactly
    xb32[idx] = cvt_pk_bits(x[2 * idx], x[2 * idx + 1]);
}

// ---------- fragment loaders ----------
static __device__ __forceinline__ v16bf load_a_frag(const unsigned short* ap) {
    // ap -> As[m][kk + khalf*16]; low 8 halves at +0, high 8 at +8
    v16bf a;
    uint4 lo = *(const uint4*)ap;
    uint4 hi = *(const uint4*)(ap + 8);
    __builtin_memcpy(&a, &lo, 16);
    __builtin_memcpy((char*)&a + 16, &hi, 16);
    return a;
}
static __device__ __forceinline__ v16bf load_b_frag_f32(const float* p) {
    const float4* q = (const float4*)p;
    unsigned u[8];
#pragma unroll
    for (int j = 0; j < 4; ++j) {
        float4 f = q[j];
        u[2 * j]     = cvt_pk_bits(f.x, f.y);
        u[2 * j + 1] = cvt_pk_bits(f.z, f.w);
    }
    v16bf b; __builtin_memcpy(&b, u, 32);
    return b;
}

// ---------- 3: expert GEMM1: h = silu(X@W1^T) * (X@W2^T) ----------
__global__ __launch_bounds__(256)
void moe_gemm1(const unsigned short* __restrict__ xb,
               const float* __restrict__ W1, const float* __restrict__ W2,
               const int* __restrict__ cnt, const int* __restrict__ tok,
               unsigned short* __restrict__ h) {
    const int e = blockIdx.z;
    const int cntE = cnt[e];
    const int m0 = blockIdx.y * MB;
    if (m0 >= cntE) return;

    const int tid = threadIdx.x;
    const int wave = tid >> 5, lane = tid & 31;
    const int col = blockIdx.x * 128 + wave * 16 + (lane & 15);  // i index
    const int khalf = lane >> 4;

    __shared__ alignas(16) unsigned short As[MB][136];
    __shared__ int rowTok[MB];

    if (tid < MB) {
        int row = m0 + tid;
        rowTok[tid] = (row < cntE) ? tok[e * N_TOK + row] : tok[e * N_TOK];
    }
    __syncthreads();

    const float* w1p = W1 + ((size_t)e * I_DIM + col) * C_DIM;
    const float* w2p = W2 + ((size_t)e * I_DIM + col) * C_DIM;

    v8f acc1[4] = {}; v8f acc2[4] = {};

    for (int k0 = 0; k0 < C_DIM; k0 += 128) {
        // stage A chunk: 64 rows x 128 bf16; 32 halves (4x b128) per thread
        {
            int r   = tid >> 2;
            int c32 = (tid & 3) * 32;
            int t   = rowTok[r] >> 1;             // token id
            const uint4* src = (const uint4*)(xb + (size_t)t * C_DIM + k0 + c32);
#pragma unroll
            for (int j = 0; j < 4; ++j)
                *(uint4*)&As[r][c32 + 8 * j] = src[j];
        }
        __syncthreads();
#pragma unroll
        for (int kk = 0; kk < 128; kk += 32) {
            v16bf b1 = load_b_frag_f32(w1p + k0 + kk + khalf * 16);
            v16bf b2 = load_b_frag_f32(w2p + k0 + kk + khalf * 16);
#pragma unroll
            for (int s = 0; s < 4; ++s) {
                v16bf a = load_a_frag(&As[s * 16 + (lane & 15)][kk + khalf * 16]);
                acc1[s] = __builtin_amdgcn_wmma_f32_16x16x32_bf16(
                              false, a, false, b1, (short)0, acc1[s], false, false);
                acc2[s] = __builtin_amdgcn_wmma_f32_16x16x32_bf16(
                              false, a, false, b2, (short)0, acc2[s], false, false);
            }
        }
        __syncthreads();
    }
    // epilogue: SwiGLU, scatter to h[entry][i]
#pragma unroll
    for (int s = 0; s < 4; ++s) {
#pragma unroll
        for (int r = 0; r < 8; ++r) {
            int mrow = m0 + s * 16 + khalf * 8 + r;
            if (mrow < cntE) {
                int entry = tok[e * N_TOK + mrow];
                float g  = acc1[s][r];
                float hv = (g / (1.0f + __expf(-g))) * acc2[s][r];
                h[(size_t)entry * I_DIM + col] = bfbits(hv);
            }
        }
    }
}

// ---------- 4: expert GEMM2: ye = H @ Wp^T ----------
__global__ __launch_bounds__(256)
void moe_gemm2(const unsigned short* __restrict__ h,
               const float* __restrict__ Wp,
               const int* __restrict__ cnt, const int* __restrict__ tok,
               float* __restrict__ ye) {
    const int e = blockIdx.z;
    const int cntE = cnt[e];
    const int m0 = blockIdx.y * MB;
    if (m0 >= cntE) return;

    const int tid = threadIdx.x;
    const int wave = tid >> 5, lane = tid & 31;
    const int col = blockIdx.x * 128 + wave * 16 + (lane & 15);  // c index
    const int khalf = lane >> 4;

    __shared__ alignas(16) unsigned short As[MB][136];
    __shared__ int rowEnt[MB];

    if (tid < MB) {
        int row = m0 + tid;
        rowEnt[tid] = (row < cntE) ? tok[e * N_TOK + row] : tok[e * N_TOK];
    }
    __syncthreads();

    const float* wpp = Wp + ((size_t)e * C_DIM + col) * I_DIM;

    v8f acc[4] = {};

    for (int k0 = 0; k0 < I_DIM; k0 += 128) {
        {
            int r   = tid >> 2;
            int c32 = (tid & 3) * 32;
            int t   = rowEnt[r];                  // entry = token*2+slot
            const uint4* src = (const uint4*)(h + (size_t)t * I_DIM + k0 + c32);
#pragma unroll
            for (int j = 0; j < 4; ++j)
                *(uint4*)&As[r][c32 + 8 * j] = src[j];
        }
        __syncthreads();
#pragma unroll
        for (int kk = 0; kk < 128; kk += 32) {
            v16bf b = load_b_frag_f32(wpp + k0 + kk + khalf * 16);
#pragma unroll
            for (int s = 0; s < 4; ++s) {
                v16bf a = load_a_frag(&As[s * 16 + (lane & 15)][kk + khalf * 16]);
                acc[s] = __builtin_amdgcn_wmma_f32_16x16x32_bf16(
                             false, a, false, b, (short)0, acc[s], false, false);
            }
        }
        __syncthreads();
    }
#pragma unroll
    for (int s = 0; s < 4; ++s) {
#pragma unroll
        for (int r = 0; r < 8; ++r) {
            int mrow = m0 + s * 16 + khalf * 8 + r;
            if (mrow < cntE) {
                int entry = tok[e * N_TOK + mrow];
                ye[(size_t)entry * C_DIM + col] = acc[s][r];
            }
        }
    }
}

// ---------- 5: weighted combine ----------
__global__ __launch_bounds__(256)
void moe_combine(const float* __restrict__ ye, const float* __restrict__ wrec,
                 float* __restrict__ y) {
    int idx = blockIdx.x * 256 + threadIdx.x;   // grid covers N*C exactly
    int n = idx / C_DIM, c = idx - n * C_DIM;
    y[idx] = wrec[2 * n]     * ye[(size_t)(2 * n)     * C_DIM + c]
           + wrec[2 * n + 1] * ye[(size_t)(2 * n + 1) * C_DIM + c];
}

// ---------- host launch ----------
extern "C" void kernel_launch(void* const* d_in, const int* in_sizes, int n_in,
                              void* d_out, int out_size, void* d_ws, size_t ws_size,
                              hipStream_t stream) {
    const float* x  = (const float*)d_in[0];
    const float* Wg = (const float*)d_in[1];
    const float* W1 = (const float*)d_in[2];
    const float* W2 = (const float*)d_in[3];
    const float* Wp = (const float*)d_in[4];
    float* y = (float*)d_out;

    char* ws = (char*)d_ws;
    int*            cnt  = (int*)(ws + 0);
    int*            tok  = (int*)(ws + 256);
    float*          wrec = (float*)(ws + 33280);
    unsigned short* xb   = (unsigned short*)(ws + 41472);
    unsigned short* h    = (unsigned short*)(ws + 4235776);
    float*          ye   = (float*)(ws + 27304448);
    (void)in_sizes; (void)n_in; (void)out_size; (void)ws_size;

    moe_zero_cnt<<<1, 32, 0, stream>>>(cnt);
    moe_router<<<N_TOK / 8, 256, 0, stream>>>(x, Wg, cnt, tok, wrec);
    moe_cvt_x<<<(N_TOK * C_DIM / 2) / 256, 256, 0, stream>>>(x, (unsigned*)xb);

    dim3 g1(I_DIM / 128, N_TOK / MB, E_NUM);
    moe_gemm1<<<g1, 256, 0, stream>>>(xb, W1, W2, cnt, tok, h);

    dim3 g2(C_DIM / 128, N_TOK / MB, E_NUM);
    moe_gemm2<<<g2, 256, 0, stream>>>(h, Wp, cnt, tok, ye);

    moe_combine<<<(N_TOK * C_DIM) / 256, 256, 0, stream>>>(ye, wrec, y);
}
